// EnhancedEmotionClassifier_50036368998671
// MI455X (gfx1250) — matrix operand order repaired and checked by reading, not compiled, not run on hardware
//
#include <hip/hip_runtime.h>
#include <hip/hip_bf16.h>

typedef __attribute__((ext_vector_type(16))) _Float16 v16h;
typedef __attribute__((ext_vector_type(8)))  _Float16 v8h;
typedef __attribute__((ext_vector_type(8)))  float    v8f;

#define EPSF 1e-5f

// ---------------- epilogue flags ----------------
constexpr int F_BIAS = 1, F_GELU = 2, F_RELU = 4, F_TANH = 8,
              F_BN = 16, F_ROWSCALE = 32, F_OUT16 = 64, F_OUT32 = 128;

__device__ __forceinline__ float gelu_exact(float x) {
  return 0.5f * x * (1.f + erff(x * 0.70710678118654752f));
}

__device__ __forceinline__ v16h load_frag(const _Float16* p0, const _Float16* p1) {
  const v8h a = *(const v8h*)p0;
  const v8h b = *(const v8h*)p1;
  v16h r;
#pragma unroll
  for (int e = 0; e < 8; ++e) { r[e] = a[e]; r[e + 8] = b[e]; }
  return r;
}

struct GemmP {
  const _Float16* A;     // activations, f16
  const _Float16* W;     // weights [N,K] row-major, f16
  const float *bias, *bng, *bnb, *bnm, *bnv, *rowscale;
  _Float16* out16;
  float*    out32;
  int K;
  int mlshift;           // b = m >> mlshift ; l = m & mask  (use 30 for flat M)
  int apad, ac;          // A row offset = (b*apad + l)*ac   (K contiguous from there)
  int cpad, chalo, cstride; // C offset = (b*cpad + l + chalo)*cstride + n
};

// C = A(MxK) @ W^T  with WMMA f32<=f16 16x16x32.
// Wave tile 32(M) x 64(N); 8 waves/block -> 64 x 256 block tile.
// Requires: M % 64 == 0, N % 256 == 0, K % 32 == 0 (all call sites satisfy this).
template <int FLAGS>
__global__ __launch_bounds__(256) void gemm_wmma(GemmP p) {
  const int lane = threadIdx.x & 31;
  const int wv   = threadIdx.x >> 5;
  const int r    = lane & 15;
  const int hi   = lane >> 4;
  const int m0   = blockIdx.x * 64  + (wv & 1) * 32;
  const int n0   = blockIdx.y * 256 + (wv >> 1) * 64;
  const int mmask = (1 << p.mlshift) - 1;

  const _Float16* abase[2];
#pragma unroll
  for (int i = 0; i < 2; ++i) {
    int m = m0 + 16 * i + r;
    int b = m >> p.mlshift, l = m & mmask;
    abase[i] = p.A + (size_t)(b * p.apad + l) * p.ac;
  }
  const _Float16* wbase[4];
#pragma unroll
  for (int j = 0; j < 4; ++j) {
    int n = n0 + 16 * j + r;
    wbase[j] = p.W + (size_t)n * p.K + 16 * hi;
  }

  v8f acc[2][4];
#pragma unroll
  for (int i = 0; i < 2; ++i)
#pragma unroll
    for (int j = 0; j < 4; ++j)
#pragma unroll
      for (int e = 0; e < 8; ++e) acc[i][j][e] = 0.f;

  for (int k = 0; k < p.K; k += 32) {
    v16h af[2], bf[4];
#pragma unroll
    for (int i = 0; i < 2; ++i)
      af[i] = load_frag(abase[i] + k + 16 * hi, abase[i] + k + 16 * hi + 8);
#pragma unroll
    for (int j = 0; j < 4; ++j)
      bf[j] = load_frag(wbase[j] + k, wbase[j] + k + 8);
    if (k + 32 < p.K) {                       // gfx1250 global_prefetch_b8
      __builtin_prefetch(abase[0] + k + 32, 0, 3);
      __builtin_prefetch(wbase[0] + k + 32, 0, 3);
    }
#pragma unroll
    for (int i = 0; i < 2; ++i)
#pragma unroll
      for (int j = 0; j < 4; ++j)
        acc[i][j] = __builtin_amdgcn_wmma_f32_16x16x32_f16(
            false, af[i], false, bf[j], (short)0, acc[i][j], false, false);
  }

#pragma unroll
  for (int i = 0; i < 2; ++i)
#pragma unroll
    for (int j = 0; j < 4; ++j) {
      const int ng = n0 + 16 * j + r;
      float bia = (FLAGS & F_BIAS) ? p.bias[ng] : 0.f;
      float sc = 1.f, sh = 0.f;
      if (FLAGS & F_BN) {
        sc = p.bng[ng] * __frsqrt_rn(p.bnv[ng] + EPSF);
        sh = p.bnb[ng] - p.bnm[ng] * sc;
      }
#pragma unroll
      for (int e = 0; e < 8; ++e) {
        int m = m0 + 16 * i + e + 8 * hi;
        float v = acc[i][j][e] + bia;
        if (FLAGS & F_GELU) v = gelu_exact(v);
        if (FLAGS & F_RELU) v = fmaxf(v, 0.f);
        if (FLAGS & F_TANH) v = tanhf(v);
        if (FLAGS & F_BN)   v = v * sc + sh;
        if (FLAGS & F_ROWSCALE) v *= p.rowscale[m & 511];
        int b = m >> p.mlshift, l = m & mmask;
        size_t co = (size_t)(b * p.cpad + l + p.chalo) * p.cstride + ng;
        if (FLAGS & F_OUT16) p.out16[co] = (_Float16)v;
        if (FLAGS & F_OUT32) p.out32[co] = v;
      }
    }
}

// ---------------- attention WMMA kernels ----------------
// S[n,head,l,m] = scale * q(l,n,head,:)·k(m,n,head,:)
// One 8-wave block per (n,head): wave tile 32(l) x 64(m); K = hd (64 or 128).
__global__ __launch_bounds__(256) void attn_qk_wmma(const _Float16* qkv, float* S,
                                                    int H, int hd, float scale) {
  const int n = blockIdx.x, head = blockIdx.y;
  const int lane = threadIdx.x & 31, wv = threadIdx.x >> 5;
  const int r = lane & 15, hi = lane >> 4;
  const int m0 = (wv & 3) * 32;       // l tile
  const int n0 = (wv >> 2) * 64;      // m tile

  const _Float16* abase[2];
#pragma unroll
  for (int i = 0; i < 2; ++i) {
    int l = m0 + 16 * i + r;
    abase[i] = qkv + ((size_t)(l * 512 + n)) * 1536 + head * hd;
  }
  const _Float16* wbase[4];
#pragma unroll
  for (int j = 0; j < 4; ++j) {
    int mm = n0 + 16 * j + r;
    wbase[j] = qkv + ((size_t)(mm * 512 + n)) * 1536 + 512 + head * hd + 16 * hi;
  }

  v8f acc[2][4];
#pragma unroll
  for (int i = 0; i < 2; ++i)
#pragma unroll
    for (int j = 0; j < 4; ++j)
#pragma unroll
      for (int e = 0; e < 8; ++e) acc[i][j][e] = 0.f;

  for (int k = 0; k < hd; k += 32) {
    v16h af[2], bf[4];
#pragma unroll
    for (int i = 0; i < 2; ++i)
      af[i] = load_frag(abase[i] + k + 16 * hi, abase[i] + k + 16 * hi + 8);
#pragma unroll
    for (int j = 0; j < 4; ++j)
      bf[j] = load_frag(wbase[j] + k, wbase[j] + k + 8);
#pragma unroll
    for (int i = 0; i < 2; ++i)
#pragma unroll
      for (int j = 0; j < 4; ++j)
        acc[i][j] = __builtin_amdgcn_wmma_f32_16x16x32_f16(
            false, af[i], false, bf[j], (short)0, acc[i][j], false, false);
  }

  float* Sbase = S + ((size_t)(n * H + head)) * 128 * 128;
#pragma unroll
  for (int i = 0; i < 2; ++i)
#pragma unroll
    for (int j = 0; j < 4; ++j) {
      int col = n0 + 16 * j + r;
#pragma unroll
      for (int e = 0; e < 8; ++e) {
        int l = m0 + 16 * i + e + 8 * hi;
        Sbase[(size_t)l * 128 + col] = acc[i][j][e] * scale;
      }
    }
}

// O[l,n,head*hd+d] = sum_m P[n,head,l,m] * Vt[n, head*hd+d, m]
// Wave tile 32(l) x 64(d); K = 128 (m). hd=64 -> 4 waves/block, hd=128 -> 8.
__global__ __launch_bounds__(256) void attn_pv_wmma(const _Float16* P, const _Float16* Vt,
                                                    _Float16* obuf, int H, int hd) {
  const int n = blockIdx.x, head = blockIdx.y;
  const int lane = threadIdx.x & 31, wv = threadIdx.x >> 5;
  const int r = lane & 15, hi = lane >> 4;
  const int m0 = (wv & 3) * 32;       // l tile
  const int n0 = (wv >> 2) * 64;      // d tile (0 when hd==64)

  const _Float16* abase[2];
#pragma unroll
  for (int i = 0; i < 2; ++i) {
    int l = m0 + 16 * i + r;
    abase[i] = P + (((size_t)(n * H + head)) * 128 + l) * 128;
  }
  const _Float16* wbase[4];
#pragma unroll
  for (int j = 0; j < 4; ++j) {
    int d = n0 + 16 * j + r;
    wbase[j] = Vt + ((size_t)(n * 512 + head * hd + d)) * 128 + 16 * hi;
  }

  v8f acc[2][4];
#pragma unroll
  for (int i = 0; i < 2; ++i)
#pragma unroll
    for (int j = 0; j < 4; ++j)
#pragma unroll
      for (int e = 0; e < 8; ++e) acc[i][j][e] = 0.f;

  for (int k = 0; k < 128; k += 32) {
    v16h af[2], bf[4];
#pragma unroll
    for (int i = 0; i < 2; ++i)
      af[i] = load_frag(abase[i] + k + 16 * hi, abase[i] + k + 16 * hi + 8);
#pragma unroll
    for (int j = 0; j < 4; ++j)
      bf[j] = load_frag(wbase[j] + k, wbase[j] + k + 8);
#pragma unroll
    for (int i = 0; i < 2; ++i)
#pragma unroll
      for (int j = 0; j < 4; ++j)
        acc[i][j] = __builtin_amdgcn_wmma_f32_16x16x32_f16(
            false, af[i], false, bf[j], (short)0, acc[i][j], false, false);
  }

#pragma unroll
  for (int i = 0; i < 2; ++i)
#pragma unroll
    for (int j = 0; j < 4; ++j) {
      int ch = head * hd + n0 + 16 * j + r;
#pragma unroll
      for (int e = 0; e < 8; ++e) {
        int l = m0 + 16 * i + e + 8 * hi;
        obuf[((size_t)(l * 512 + n)) * 512 + ch] = (_Float16)acc[i][j][e];
      }
    }
}

// LDS-tiled transpose: Vt[n, ch, m] = qkv[(m*512+n)*1536 + 1024 + ch]
__global__ void v_transpose(const _Float16* qkv, _Float16* Vt) {
  __shared__ _Float16 t[32][33];
  int n = blockIdx.x, mt = blockIdx.y * 32, ct = blockIdx.z * 32;
  int tx = threadIdx.x, ty = threadIdx.y;   // 32 x 8
  for (int yy = ty; yy < 32; yy += 8) {
    int m_ = mt + yy, ch = ct + tx;
    t[yy][tx] = qkv[((size_t)(m_ * 512 + n)) * 1536 + 1024 + ch];
  }
  __syncthreads();
  for (int yy = ty; yy < 32; yy += 8) {
    int ch = ct + yy, m_ = mt + tx;
    Vt[((size_t)(n * 512 + ch)) * 128 + m_] = t[tx][yy];
  }
}

// ---------------- helper kernels ----------------
__global__ void cvt16(const float* s, _Float16* d, int n) {
  int i = blockIdx.x * 256 + threadIdx.x;
  if (i < n) d[i] = (_Float16)s[i];
}

// conv w (O,I,3) -> [o, kw*I + i] f16
__global__ void repack_conv(const float* s, _Float16* d, int O, int I) {
  int idx = blockIdx.x * 256 + threadIdx.x;
  if (idx >= O * I * 3) return;
  int kw = idx % 3, i = (idx / 3) % I, o = idx / (3 * I);
  d[(size_t)o * (3 * I) + kw * I + i] = (_Float16)s[idx];
}

__global__ void init_cov(float* cov) {
  int i = blockIdx.x * 256 + threadIdx.x;
  if (i < 512) cov[i] = (i < 5) ? 1.f : 2.f;
}

// fe1: (128,193)@(1024,193)^T + bias -> gelu -> bn1 -> f16
__global__ void fe1_kernel(const float* x, const float* w, const float* bias,
                           const float* g, const float* bb, const float* mm,
                           const float* vv, _Float16* out) {
  int idx = blockIdx.x * 256 + threadIdx.x;   // 128*1024
  int row = idx >> 10, col = idx & 1023;
  const float* xr = x + row * 193;
  const float* wr = w + col * 193;
  float s = bias[col];
  for (int k = 0; k < 193; ++k) s += xr[k] * wr[k];
  s = gelu_exact(s);
  float sc = g[col] * __frsqrt_rn(vv[col] + EPSF);
  s = (s - mm[col]) * sc + bb[col];
  out[idx] = (_Float16)s;
}

// conv1 (Cin=1, k=5, pad=2) + gelu + cbn1 -> padded act1 (B,514,128) f16 (halo=0)
__global__ void conv1_kernel(const float* h2, const float* w, const float* bias,
                             const float* g, const float* bb, const float* mm,
                             const float* vv, _Float16* act1) {
  int idx = blockIdx.x * 256 + threadIdx.x;   // 128*514*128
  int c = idx & 127;
  int lp = (idx >> 7) % 514;
  int b = idx / (514 * 128);
  if (lp == 0 || lp == 513) { act1[idx] = (_Float16)0.f; return; }
  int l = lp - 1;
  float s = bias[c];
#pragma unroll
  for (int kw = 0; kw < 5; ++kw) {
    int li = l + kw - 2;
    if (li >= 0 && li < 512) s += w[c * 5 + kw] * h2[b * 512 + li];
  }
  s = gelu_exact(s);
  float sc = g[c] * __frsqrt_rn(vv[c] + EPSF);
  s = (s - mm[c]) * sc + bb[c];
  act1[idx] = (_Float16)s;
}

__global__ void zero_act2_halo(_Float16* act2) {
  int idx = blockIdx.x * 256 + threadIdx.x;   // 128*2*256
  int c = idx & 255, which = (idx >> 8) & 1, b = idx >> 9;
  act2[((size_t)b * 514 + (which ? 513 : 0)) * 256 + c] = (_Float16)0.f;
}

// softmax over the 128-wide last axis; one wave32 per row; emits f16 P
__global__ void attn_softmax(const float* S, _Float16* P, int rows) {
  int gw = (blockIdx.x * blockDim.x + threadIdx.x) >> 5;
  int lane = threadIdx.x & 31;
  if (gw >= rows) return;
  const float* row = S + (size_t)gw * 128;
  float v[4], mx = -3.4e38f;
#pragma unroll
  for (int i = 0; i < 4; ++i) { v[i] = row[lane + 32 * i]; mx = fmaxf(mx, v[i]); }
#pragma unroll
  for (int off = 16; off >= 1; off >>= 1) mx = fmaxf(mx, __shfl_xor(mx, off, 32));
  float sum = 0.f;
#pragma unroll
  for (int i = 0; i < 4; ++i) { v[i] = expf(v[i] - mx); sum += v[i]; }
#pragma unroll
  for (int off = 16; off >= 1; off >>= 1) sum += __shfl_xor(sum, off, 32);
  float inv = 1.f / sum;
  _Float16* prow = P + (size_t)gw * 128;
#pragma unroll
  for (int i = 0; i < 4; ++i) prow[lane + 32 * i] = (_Float16)(v[i] * inv);
}

// ea2: wraw[row] = tanh_out[row,:256] · ea2_w + ea2_b ; one wave per row
__global__ void ea2_kernel(const float* t, const float* w2, const float* b2, float* wraw) {
  int gw = (blockIdx.x * blockDim.x + threadIdx.x) >> 5;
  int lane = threadIdx.x & 31;
  const float* row = t + (size_t)gw * 256;
  float s = 0.f;
  for (int k = lane; k < 256; k += 32) s += row[k] * w2[k];
#pragma unroll
  for (int off = 16; off >= 1; off >>= 1) s += __shfl_xor(s, off, 32);
  if (lane == 0) wraw[gw] = s + b2[0];
}

// softmax over n (512) per batch row; writes final attention weights
__global__ void softmax_w(const float* wraw, float* wout) {
  __shared__ float sm[256];
  int b = blockIdx.x, t = threadIdx.x;
  float a0 = wraw[b * 512 + t], a1 = wraw[b * 512 + t + 256];
  sm[t] = fmaxf(a0, a1);
  __syncthreads();
  for (int s = 128; s > 0; s >>= 1) { if (t < s) sm[t] = fmaxf(sm[t], sm[t + s]); __syncthreads(); }
  float mx = sm[0];
  __syncthreads();
  float e0 = expf(a0 - mx), e1 = expf(a1 - mx);
  sm[t] = e0 + e1;
  __syncthreads();
  for (int s = 128; s > 0; s >>= 1) { if (t < s) sm[t] += sm[t + s]; __syncthreads(); }
  float inv = 1.f / sm[0];
  wout[b * 512 + t] = e0 * inv;
  wout[b * 512 + t + 256] = e1 * inv;
}

// att[b,c] = sum_n hfus[b,n,c] * w[b,n]
__global__ void att_kernel(const float* hfus, const float* w, float* att) {
  __shared__ float ws[512];
  int b = blockIdx.x, c = threadIdx.x;        // 512 threads
  ws[c] = w[b * 512 + c];
  __syncthreads();
  const float* hb = hfus + (size_t)b * 512 * 512 + c;
  float s = 0.f;
  for (int n = 0; n < 512; ++n) s += hb[(size_t)n * 512] * ws[n];
  att[b * 512 + c] = s;
}

// path + pbn -> cf (128,1024)
__global__ void path_kernel(const float* att, const float* pw, const float* pb,
                            const float* g, const float* bb, const float* mm,
                            const float* vv, float* cf) {
  int idx = blockIdx.x * 256 + threadIdx.x;   // 128*1024
  int b = idx >> 10, ko = idx & 1023;
  const float* a = att + b * 512;
  const float* wr = pw + (size_t)ko * 512;
  float s = pb[ko];
  for (int d = 0; d < 512; ++d) s += a[d] * wr[d];
  s = gelu_exact(s);
  float sc = g[ko] * __frsqrt_rn(vv[ko] + EPSF);
  s = (s - mm[ko]) * sc + bb[ko];
  cf[idx] = s;
}

__global__ void heads_kernel(const float* cf, const float* ew, const float* eb,
                             const float* vw, const float* vb, const float* aw,
                             const float* ab, const float* lw, const float* lb,
                             float* out) {
  int b = blockIdx.x, j = threadIdx.x;
  if (j >= 13) return;
  const float* c = cf + b * 1024;
  const float* wr; float bias; float* dst;
  if (j < 8)       { wr = ew + j * 1024;        bias = eb[j];     dst = out + b * 8 + j; }
  else if (j < 11) { int q = j - 8; wr = vw + q * 1024; bias = vb[q]; dst = out + 1024 + b * 3 + q; }
  else if (j == 11){ wr = aw; bias = ab[0]; dst = out + 1408 + b; }
  else             { wr = lw; bias = lb[0]; dst = out + 1536 + b; }
  float s = bias;
  for (int k = 0; k < 1024; ++k) s += c[k] * wr[k];
  *dst = s;
}

// ---------------- launcher ----------------
extern "C" void kernel_launch(void* const* d_in, const int* in_sizes, int n_in,
                              void* d_out, int out_size, void* d_ws, size_t ws_size,
                              hipStream_t stream) {
  (void)in_sizes; (void)n_in; (void)out_size; (void)ws_size;
  const float* x       = (const float*)d_in[0];
  const float* fe1_w   = (const float*)d_in[1],  *fe1_b = (const float*)d_in[2];
  const float* bn1g = (const float*)d_in[3], *bn1b = (const float*)d_in[4],
             * bn1m = (const float*)d_in[5], *bn1v = (const float*)d_in[6];
  const float* fe2_w   = (const float*)d_in[7],  *fe2_b = (const float*)d_in[8];
  const float* bn2g = (const float*)d_in[9], *bn2b = (const float*)d_in[10],
             * bn2m = (const float*)d_in[11], *bn2v = (const float*)d_in[12];
  const float* c1w = (const float*)d_in[13], *c1b = (const float*)d_in[14];
  const float* cb1g=(const float*)d_in[15],*cb1b=(const float*)d_in[16],
             * cb1m=(const float*)d_in[17],*cb1v=(const float*)d_in[18];
  const float* c2w = (const float*)d_in[19], *c2b = (const float*)d_in[20];
  const float* cb2g=(const float*)d_in[21],*cb2b=(const float*)d_in[22],
             * cb2m=(const float*)d_in[23],*cb2v=(const float*)d_in[24];
  const float* c3w = (const float*)d_in[25], *c3b = (const float*)d_in[26];
  const float* cb3g=(const float*)d_in[27],*cb3b=(const float*)d_in[28],
             * cb3m=(const float*)d_in[29],*cb3v=(const float*)d_in[30];
  const float* ga_inw=(const float*)d_in[31], *ga_inb=(const float*)d_in[32];
  const float* ga_ow =(const float*)d_in[33], *ga_ob =(const float*)d_in[34];
  const float* la_inw=(const float*)d_in[35], *la_inb=(const float*)d_in[36];
  const float* la_ow =(const float*)d_in[37], *la_ob =(const float*)d_in[38];
  const float* fus_w =(const float*)d_in[39], *fus_b =(const float*)d_in[40];
  const float* ea1_w =(const float*)d_in[41], *ea1_b =(const float*)d_in[42];
  const float* ea2_w =(const float*)d_in[43], *ea2_b =(const float*)d_in[44];
  const float* path_w=(const float*)d_in[45], *path_b=(const float*)d_in[46];
  const float* pbng=(const float*)d_in[47],*pbnb=(const float*)d_in[48],
             * pbnm=(const float*)d_in[49],*pbnv=(const float*)d_in[50];
  const float* emo_w=(const float*)d_in[51],*emo_b=(const float*)d_in[52];
  const float* vad_w=(const float*)d_in[53],*vad_b=(const float*)d_in[54];
  const float* aro_w=(const float*)d_in[55],*aro_b=(const float*)d_in[56];
  const float* val_w=(const float*)d_in[57],*val_b=(const float*)d_in[58];
  float* out = (float*)d_out;

  char* wsb = (char*)d_ws; size_t off = 0;
  auto alloc = [&](size_t bytes) -> void* {
    void* p = wsb + off;
    off = (off + bytes + 255) & ~(size_t)255;
    return p;
  };
  _Float16* wfe2   = (_Float16*)alloc((size_t)512*1024*2);
  _Float16* wc2    = (_Float16*)alloc((size_t)256*384*2);
  _Float16* wc3    = (_Float16*)alloc((size_t)512*768*2);
  _Float16* wgain  = (_Float16*)alloc((size_t)1536*512*2);
  _Float16* wgaout = (_Float16*)alloc((size_t)512*512*2);
  _Float16* wlain  = (_Float16*)alloc((size_t)1536*512*2);
  _Float16* wlaout = (_Float16*)alloc((size_t)512*512*2);
  _Float16* wfus   = (_Float16*)alloc((size_t)512*1024*2);
  _Float16* wea1   = (_Float16*)alloc((size_t)256*512*2);
  _Float16* h1     = (_Float16*)alloc((size_t)128*1024*2);
  float*    h2     = (float*)   alloc((size_t)128*512*4);
  _Float16* act1   = (_Float16*)alloc((size_t)128*514*128*2);
  _Float16* act2   = (_Float16*)alloc((size_t)128*514*256*2);
  _Float16* hbuf   = (_Float16*)alloc((size_t)65536*512*2);
  _Float16* qkv    = (_Float16*)alloc((size_t)65536*1536*2);
  float*    S      = (float*)   alloc((size_t)512*8*128*128*4);
  _Float16* P16    = (_Float16*)alloc((size_t)512*8*128*128*2);
  _Float16* Vt     = (_Float16*)alloc((size_t)512*512*128*2);
  _Float16* obuf   = (_Float16*)alloc((size_t)65536*512*2);
  _Float16* comb   = (_Float16*)alloc((size_t)65536*1024*2);
  _Float16* hfus16 = (_Float16*)alloc((size_t)65536*512*2);
  float*    hfus32 = (float*)   alloc((size_t)65536*512*4);
  float*    ea1o   = (float*)   alloc((size_t)65536*256*4);
  float*    cov    = (float*)   alloc(512*4);
  float*    wraw   = (float*)   alloc(65536*4);
  float*    attb   = (float*)   alloc(65536*4);
  float*    cf     = (float*)   alloc(131072*4);

  auto cv = [&](const float* s, _Float16* d, int n) {
    cvt16<<<(n + 255) / 256, 256, 0, stream>>>(s, d, n);
  };
  cv(fe2_w, wfe2, 512*1024);
  repack_conv<<<(256*128*3 + 255)/256, 256, 0, stream>>>(c2w, wc2, 256, 128);
  repack_conv<<<(512*256*3 + 255)/256, 256, 0, stream>>>(c3w, wc3, 512, 256);
  cv(ga_inw, wgain, 1536*512);  cv(ga_ow, wgaout, 512*512);
  cv(la_inw, wlain, 1536*512);  cv(la_ow, wlaout, 512*512);
  cv(fus_w, wfus, 512*1024);    cv(ea1_w, wea1, 256*512);
  init_cov<<<2, 256, 0, stream>>>(cov);

  // fe1 (fp32, K=193) + gelu + bn1 -> f16
  fe1_kernel<<<(128*1024)/256, 256, 0, stream>>>(x, fe1_w, fe1_b, bn1g, bn1b, bn1m, bn1v, h1);

  // fe2: 128x512x1024 WMMA, gelu+bn2 -> f32
  {
    GemmP p{h1, wfe2, fe2_b, bn2g, bn2b, bn2m, bn2v, nullptr, nullptr, h2,
            1024, 30, 0, 1024, 0, 0, 512};
    gemm_wmma<F_BIAS|F_GELU|F_BN|F_OUT32><<<dim3(2,2), 256, 0, stream>>>(p);
  }
  // conv1 (fp32) -> padded act1
  conv1_kernel<<<(128*514*128)/256, 256, 0, stream>>>(h2, c1w, c1b, cb1g, cb1b, cb1m, cb1v, act1);
  // conv2 as GEMM 65536x256x384 -> padded act2
  {
    GemmP p{act1, wc2, c2b, cb2g, cb2b, cb2m, cb2v, nullptr, act2, nullptr,
            384, 9, 514, 128, 514, 1, 256};
    gemm_wmma<F_BIAS|F_GELU|F_BN|F_OUT16><<<dim3(1024,1), 256, 0, stream>>>(p);
  }
  zero_act2_halo<<<(128*2*256)/256, 256, 0, stream>>>(act2);
  // conv3 as GEMM 65536x512x768 -> h (B,L,C) f16 (already "transposed" layout)
  {
    GemmP p{act2, wc3, c3b, cb3g, cb3b, cb3m, cb3v, nullptr, hbuf, nullptr,
            768, 9, 514, 256, 512, 0, 512};
    gemm_wmma<F_BIAS|F_GELU|F_BN|F_OUT16><<<dim3(1024,2), 256, 0, stream>>>(p);
  }

  // ---- two attention branches, sequentially reusing qkv/S/P16/Vt/obuf ----
  for (int branch = 0; branch < 2; ++branch) {
    const bool ga = (branch == 0);
    const _Float16* win = ga ? wgain : wlain;
    const float* binb   = ga ? ga_inb : la_inb;
    const _Float16* wo  = ga ? wgaout : wlaout;
    const float* bob    = ga ? ga_ob  : la_ob;
    const int H = ga ? 8 : 4, hd = ga ? 64 : 128;
    const float scale = ga ? 0.125f : 0.08838834764831845f; // 1/sqrt(hd)

    GemmP pq{hbuf, win, binb, nullptr, nullptr, nullptr, nullptr, nullptr, qkv, nullptr,
             512, 30, 0, 512, 0, 0, 1536};
    gemm_wmma<F_BIAS|F_OUT16><<<dim3(1024,6), 256, 0, stream>>>(pq);

    attn_qk_wmma<<<dim3(512, H), 256, 0, stream>>>(qkv, S, H, hd, scale);
    int rows = 512 * H * 128;
    attn_softmax<<<rows / 8, 256, 0, stream>>>(S, P16, rows);
    v_transpose<<<dim3(512, 4, 16), dim3(32, 8), 0, stream>>>(qkv, Vt);
    attn_pv_wmma<<<dim3(512, H), (hd == 128 ? 256 : 128), 0, stream>>>(P16, Vt, obuf, H, hd);

    if (ga) {
      GemmP po{obuf, wo, bob, nullptr, nullptr, nullptr, nullptr, nullptr, comb, nullptr,
               512, 30, 0, 512, 0, 0, 1024};
      gemm_wmma<F_BIAS|F_OUT16><<<dim3(1024,2), 256, 0, stream>>>(po);
    } else {
      GemmP po{obuf, wo, bob, nullptr, nullptr, nullptr, nullptr, cov, comb + 512, nullptr,
               512, 30, 0, 512, 0, 0, 1024};
      gemm_wmma<F_BIAS|F_ROWSCALE|F_OUT16><<<dim3(1024,2), 256, 0, stream>>>(po);
    }
  }

  // fus: 65536x512x1024, relu -> f16 + f32
  {
    GemmP p{comb, wfus, fus_b, nullptr, nullptr, nullptr, nullptr, nullptr, hfus16, hfus32,
            1024, 30, 0, 1024, 0, 0, 512};
    gemm_wmma<F_BIAS|F_RELU|F_OUT16|F_OUT32><<<dim3(1024,2), 256, 0, stream>>>(p);
  }
  // ea1: 65536x256x512, tanh -> f32
  {
    GemmP p{hfus16, wea1, ea1_b, nullptr, nullptr, nullptr, nullptr, nullptr, nullptr, ea1o,
            512, 30, 0, 512, 0, 0, 256};
    gemm_wmma<F_BIAS|F_TANH|F_OUT32><<<dim3(1024,1), 256, 0, stream>>>(p);
  }
  ea2_kernel<<<8192, 256, 0, stream>>>(ea1o, ea2_w, ea2_b, wraw);
  softmax_w<<<128, 256, 0, stream>>>(wraw, out + 1664);       // final w output
  att_kernel<<<128, 512, 0, stream>>>(hfus32, out + 1664, attb);
  path_kernel<<<(128*1024)/256, 256, 0, stream>>>(attb, path_w, path_b,
                                                  pbng, pbnb, pbnm, pbnv, cf);
  heads_kernel<<<128, 32, 0, stream>>>(cf, emo_w, emo_b, vad_w, vad_b,
                                       aro_w, aro_b, val_w, val_b, out);
}
